// LowRank_Attention_55009941127554
// MI455X (gfx1250) — matrix operand order repaired and checked by reading, not compiled, not run on hardware
//
#include <hip/hip_runtime.h>
#include <hip/hip_bf16.h>

// ---------------------------------------------------------------------------
// Performer / low-rank linear attention on gfx1250 (MI455X), bf16 WMMA path.
// All WMMA operands are laid out K-contiguous so every fragment is built from
// two 16-byte loads (no v_mov_b16 half-packing).
// ---------------------------------------------------------------------------

typedef __bf16 bf16;
typedef __attribute__((ext_vector_type(16))) __bf16 v16bf;
typedef __attribute__((ext_vector_type(8)))  __bf16 v8bf;
typedef __attribute__((ext_vector_type(8)))  float  v8f;

constexpr int B_  = 4;
constexpr int S_  = 8192;
constexpr int H_  = 768;
constexpr int NH_ = 12;
constexpr int HD_ = 64;
constexpr int M_  = 64;
constexpr int BS  = B_ * S_;     // 32768 rows
constexpr int N3  = 3 * H_;      // 2304 (q|k|v concat)
constexpr float EPS_ = 1e-6f;

__device__ __forceinline__ int lane_id() { return (int)(threadIdx.x & 31); }

__device__ __forceinline__ v16bf combine16(v8bf lo, v8bf hi) {
  return __builtin_shufflevector(lo, hi, 0, 1, 2, 3, 4, 5, 6, 7, 8, 9, 10, 11,
                                 12, 13, 14, 15);
}

// A fragment 16x32 (MxK), K-contiguous source: element(r,k) = base[r*rs + k].
// lane L: r=L&15, hi=L>>4, holds K={hi*8..+7} then {16+hi*8..+7}.
__device__ __forceinline__ v16bf load_a_frag(const bf16* base, int rs) {
  const int l = lane_id(), r = l & 15, hi = l >> 4;
  const bf16* p = base + (size_t)r * rs + hi * 8;
  return combine16(*(const v8bf*)p, *(const v8bf*)(p + 16));
}

// B fragment 32x16 (KxN) from N-major/K-contiguous (i.e. transposed) source:
// element(k,n) = base[n*ns + k].  lane L: n=L&15, hi=L>>4, holds K={hi*8..+7} then {16+hi*8..+7}.
__device__ __forceinline__ v16bf load_b_frag(const bf16* base, int ns) {
  const int l = lane_id(), n = l & 15, hi = l >> 4;
  const bf16* p = base + (size_t)n * ns + hi * 8;
  return combine16(*(const v8bf*)p, *(const v8bf*)(p + 16));
}

__device__ __forceinline__ v8f wmma_bf16(v16bf a, v16bf b, v8f c) {
  return __builtin_amdgcn_wmma_f32_16x16x32_bf16(false, a, false, b,
                                                 (short)0, c, false, false);
}

// ---------------------------------------------------------------------------
// Elementwise / packing helpers
// ---------------------------------------------------------------------------
__global__ void zero_kernel(float* __restrict__ p, size_t n) {
  size_t i = (size_t)blockIdx.x * blockDim.x + threadIdx.x;
  size_t stride = (size_t)gridDim.x * blockDim.x;
  for (; i < n; i += stride) p[i] = 0.0f;
}

__global__ void cvt_kernel(const float* __restrict__ in, bf16* __restrict__ out,
                           size_t n) {
  size_t i = (size_t)blockIdx.x * blockDim.x + threadIdx.x;
  size_t stride = (size_t)gridDim.x * blockDim.x;
  for (; i < n; i += stride) out[i] = (bf16)in[i];
}

// WoT [768][768]: WoT[n][k] = Wo[k][n]
__global__ void cvtT768_kernel(const float* __restrict__ W,
                               bf16* __restrict__ WT) {
  size_t i = (size_t)blockIdx.x * blockDim.x + threadIdx.x;
  if (i >= (size_t)H_ * H_) return;
  int n = (int)(i / H_), k = (int)(i % H_);
  WT[i] = (bf16)W[(size_t)k * H_ + n];
}

// PT [NH][M][HD]: PT[h][m][d] = P[h][d][m]
__global__ void cvtPT_kernel(const float* __restrict__ P,
                             bf16* __restrict__ PT) {
  size_t i = (size_t)blockIdx.x * blockDim.x + threadIdx.x;
  if (i >= (size_t)NH_ * M_ * HD_) return;
  int h = (int)(i / (M_ * HD_));
  int rem = (int)(i % (M_ * HD_));
  int m = rem / HD_, d = rem % HD_;
  PT[i] = (bf16)P[(size_t)h * HD_ * M_ + (size_t)d * M_ + m];
}

// kvT [48][HD][M]: kvT[bh][d][m] = kv_acc[bh][m][d]
__global__ void cvtKVT_kernel(const float* __restrict__ kv,
                              bf16* __restrict__ kvT) {
  size_t i = (size_t)blockIdx.x * blockDim.x + threadIdx.x;
  if (i >= (size_t)B_ * NH_ * M_ * HD_) return;
  int bh = (int)(i / (M_ * HD_));
  int rem = (int)(i % (M_ * HD_));
  int d = rem / M_, m = rem % M_;
  kvT[i] = (bf16)kv[(size_t)bh * M_ * HD_ + (size_t)m * HD_ + d];
}

// WqkvT [2304][768] = [Wq|Wk|Wv]^T in bf16, plus bqkv [2304] fp32.
__global__ void pack_wqkvT_kernel(const float* __restrict__ Wq,
                                  const float* __restrict__ Wk,
                                  const float* __restrict__ Wv,
                                  const float* __restrict__ bq,
                                  const float* __restrict__ bk,
                                  const float* __restrict__ bv,
                                  bf16* __restrict__ WT,
                                  float* __restrict__ bias) {
  size_t i = (size_t)blockIdx.x * blockDim.x + threadIdx.x;
  if (i >= (size_t)N3 * H_) return;
  int n = (int)(i / H_), k = (int)(i % H_);
  int c = n % H_;
  const float* src = (n < H_) ? Wq : (n < 2 * H_) ? Wk : Wv;
  WT[i] = (bf16)src[(size_t)k * H_ + c];
  if (k == 0) {
    const float* bs = (n < H_) ? bq : (n < 2 * H_) ? bk : bv;
    bias[n] = bs[c];
  }
}

// ---------------------------------------------------------------------------
// Big tiled GEMM: C[M,N] = A[M,K](bf16, row-major) @ BT[N,K]^T + bias.
// WG = 256 threads (8 waves), tile 128x128, wave tile 32x64, K step 32.
// ---------------------------------------------------------------------------
template <bool OUT_BF16>
__global__ __launch_bounds__(256) void gemm_kernel(
    const bf16* __restrict__ A, const bf16* __restrict__ BT,
    const float* __restrict__ bias, void* __restrict__ Cout, int N, int K) {
  __shared__ bf16 As[128][32];   // 8 KB, M x K
  __shared__ bf16 Bs[128][32];   // 8 KB, N x K (transposed tile)
  const int m0 = blockIdx.y * 128;
  const int n0 = blockIdx.x * 128;
  const int t = threadIdx.x;
  const int wid = t >> 5;
  const int wm = (wid & 3) * 32;
  const int wn = (wid >> 2) * 64;

  v8f acc[2][4];
#pragma unroll
  for (int mt = 0; mt < 2; ++mt)
#pragma unroll
    for (int nt = 0; nt < 4; ++nt)
#pragma unroll
      for (int e = 0; e < 8; ++e) acc[mt][nt][e] = 0.0f;

  for (int k0 = 0; k0 < K; k0 += 32) {
    __syncthreads();
#pragma unroll
    for (int i = 0; i < 2; ++i) {
      int idx = t + i * 256;               // 0..511
      int row = idx >> 2;                  // 0..127
      int col = (idx & 3) * 8;             // 0,8,16,24
      *(uint4*)&As[row][col] =
          *(const uint4*)&A[(size_t)(m0 + row) * K + k0 + col];
      *(uint4*)&Bs[row][col] =
          *(const uint4*)&BT[(size_t)(n0 + row) * K + k0 + col];
    }
    __syncthreads();

    v16bf af[2], bfr[4];
    af[0] = load_a_frag(&As[wm][0], 32);
    af[1] = load_a_frag(&As[wm + 16][0], 32);
#pragma unroll
    for (int nt = 0; nt < 4; ++nt) bfr[nt] = load_b_frag(&Bs[wn + nt * 16][0], 32);
#pragma unroll
    for (int nt = 0; nt < 4; ++nt)
#pragma unroll
      for (int mt = 0; mt < 2; ++mt)
        acc[mt][nt] = wmma_bf16(af[mt], bfr[nt], acc[mt][nt]);
  }

  const int l = lane_id(), cn = l & 15, hi = l >> 4;
#pragma unroll
  for (int mt = 0; mt < 2; ++mt)
#pragma unroll
    for (int nt = 0; nt < 4; ++nt)
#pragma unroll
      for (int j = 0; j < 8; ++j) {
        int row = m0 + wm + mt * 16 + hi * 8 + j;
        int col = n0 + wn + nt * 16 + cn;
        float v = acc[mt][nt][j] + bias[col];
        if (OUT_BF16)
          ((bf16*)Cout)[(size_t)row * N + col] = (bf16)v;
        else
          ((float*)Cout)[(size_t)row * N + col] = v;
      }
}

// ---------------------------------------------------------------------------
// Feature map: qp = exp(q@P - rowmax) row-major [bh][S][M],
//              kpT = exp(k@P - rowmax) transposed [bh][M][S], ksum += columns.
// Grid (B*NH, S/64); 128 threads = 4 waves; each wave owns 16 rows.
// ---------------------------------------------------------------------------
__global__ __launch_bounds__(128) void feature_kernel(
    const bf16* __restrict__ Y,    // [BS, 2304] q|k|v
    const bf16* __restrict__ PT,   // [NH][M][HD] (K=d contiguous)
    bf16* __restrict__ qp, bf16* __restrict__ kpT, float* __restrict__ ksum) {
  const int bh = blockIdx.x, st = blockIdx.y;
  const int h = bh % NH_, b = bh / NH_;
  const int wid = threadIdx.x >> 5;
  const int l = lane_id(), cn = l & 15, hi = l >> 4;
  const size_t yrow = (size_t)b * S_ + (size_t)st * 64 + wid * 16;
  const int srow0 = st * 64 + wid * 16;

#pragma unroll
  for (int p = 0; p < 2; ++p) {   // p=0: q, p=1: k
    const bf16* Abase = Y + yrow * N3 + (p ? H_ : 0) + h * HD_;
    v8f acc[4];
#pragma unroll
    for (int nt = 0; nt < 4; ++nt)
#pragma unroll
      for (int e = 0; e < 8; ++e) acc[nt][e] = 0.0f;

#pragma unroll
    for (int ks = 0; ks < 2; ++ks) {
      v16bf a = load_a_frag(Abase + ks * 32, N3);
#pragma unroll
      for (int nt = 0; nt < 4; ++nt) {
        v16bf bfr = load_b_frag(
            PT + (size_t)h * M_ * HD_ + (size_t)(nt * 16) * HD_ + ks * 32, HD_);
        acc[nt] = wmma_bf16(a, bfr, acc[nt]);
      }
    }

    // rowmax over 64 columns (4 accumulators + 16-lane xor reduce)
    float rm[8];
#pragma unroll
    for (int j = 0; j < 8; ++j) {
      float m = fmaxf(fmaxf(acc[0][j], acc[1][j]), fmaxf(acc[2][j], acc[3][j]));
      m = fmaxf(m, __shfl_xor(m, 1, 32));
      m = fmaxf(m, __shfl_xor(m, 2, 32));
      m = fmaxf(m, __shfl_xor(m, 4, 32));
      m = fmaxf(m, __shfl_xor(m, 8, 32));
      rm[j] = m;
    }

    float csum[4] = {0.f, 0.f, 0.f, 0.f};
#pragma unroll
    for (int nt = 0; nt < 4; ++nt) {
      v8bf pk;
#pragma unroll
      for (int j = 0; j < 8; ++j) {
        float e = expf(acc[nt][j] - rm[j]);
        csum[nt] += e;
        pk[j] = (bf16)e;
        if (p == 0) {
          int row = hi * 8 + j;
          qp[((size_t)bh * S_ + srow0 + row) * M_ + nt * 16 + cn] = pk[j];
        }
      }
      if (p == 1) {
        // rows j are consecutive along S: one aligned 16B store
        *(v8bf*)&kpT[((size_t)bh * M_ + nt * 16 + cn) * S_ + srow0 + hi * 8] = pk;
      }
    }
    if (p == 1) {
#pragma unroll
      for (int nt = 0; nt < 4; ++nt)
        atomicAdd(&ksum[(size_t)bh * M_ + nt * 16 + cn], csum[nt]);
    }
  }
}

// ---------------------------------------------------------------------------
// vT [bh][HD][S]: vT[bh][d][s] = v[b][s][h*64+d]  (makes kv GEMM K-contiguous)
// Grid (48, S/64); 128 threads: lane spans d, 2 row-groups of 32.
// ---------------------------------------------------------------------------
__global__ __launch_bounds__(128) void vT_kernel(const bf16* __restrict__ Y,
                                                 bf16* __restrict__ vT) {
  const int bh = blockIdx.x, st = blockIdx.y;
  const int h = bh % NH_, b = bh / NH_;
  const int t = threadIdx.x;
  const int wid = t >> 5, l = t & 31;
  const int d = (wid & 1) * 32 + l;        // 0..63
  const int rbase = st * 64 + (wid >> 1) * 32;

  bf16 tmp[32];
#pragma unroll
  for (int i = 0; i < 32; ++i)
    tmp[i] = Y[((size_t)b * S_ + rbase + i) * N3 + 2 * H_ + h * HD_ + d];

  bf16* dst = &vT[((size_t)bh * HD_ + d) * S_ + rbase];
#pragma unroll
  for (int c = 0; c < 4; ++c) {
    v8bf pk;
#pragma unroll
    for (int j = 0; j < 8; ++j) pk[j] = tmp[c * 8 + j];
    *(v8bf*)(dst + c * 8) = pk;
  }
}

// ---------------------------------------------------------------------------
// kv[m,d] = sum_n kp[n,m]*v[n,d] per (b,h). A = kpT (K=s contig),
// B = vT (K=s contig). Grid (48, 8); fp32 atomic accumulate.
// ---------------------------------------------------------------------------
__global__ __launch_bounds__(128) void kv_kernel(const bf16* __restrict__ kpT,
                                                 const bf16* __restrict__ vT,
                                                 float* __restrict__ kv_acc) {
  const int bh = blockIdx.x, chunk = blockIdx.y;
  const int wid = threadIdx.x >> 5;
  const int mbase = wid * 16;
  const int sbase = chunk * (S_ / 8);

  v8f acc[4];
#pragma unroll
  for (int nt = 0; nt < 4; ++nt)
#pragma unroll
    for (int e = 0; e < 8; ++e) acc[nt][e] = 0.0f;

  const bf16* Ab = kpT + ((size_t)bh * M_ + mbase) * S_ + sbase;
  const bf16* Bb = vT + (size_t)bh * HD_ * S_ + sbase;
  for (int kk = 0; kk < S_ / 8; kk += 32) {
    v16bf a = load_a_frag(Ab + kk, S_);
#pragma unroll
    for (int nt = 0; nt < 4; ++nt)
      acc[nt] =
          wmma_bf16(a, load_b_frag(Bb + (size_t)(nt * 16) * S_ + kk, S_), acc[nt]);
  }

  const int l = lane_id(), cn = l & 15, hi = l >> 4;
#pragma unroll
  for (int nt = 0; nt < 4; ++nt)
#pragma unroll
    for (int j = 0; j < 8; ++j) {
      int m = mbase + hi * 8 + j;
      atomicAdd(&kv_acc[(size_t)bh * (M_ * HD_) + (size_t)m * HD_ + nt * 16 + cn],
                acc[nt][j]);
    }
}

// ---------------------------------------------------------------------------
// ctx = (qp @ kv) / (qp . k_sum + eps). A = qp (K=m contig), B = kvT (K=m
// contig). Grid (48, S/64); output bf16 [BS, 768] row-major.
// ---------------------------------------------------------------------------
__global__ __launch_bounds__(128) void ctx_kernel(
    const bf16* __restrict__ qp, const bf16* __restrict__ kvT,
    const float* __restrict__ ksum, bf16* __restrict__ ctx) {
  __shared__ float norm[64];
  const int bh = blockIdx.x, st = blockIdx.y;
  const int h = bh % NH_, b = bh / NH_;
  const int wid = threadIdx.x >> 5;
  const int l = lane_id(), cn = l & 15, hi = l >> 4;
  const size_t qrow = (size_t)bh * S_ + (size_t)st * 64 + wid * 16;

  v8f acc[4];
#pragma unroll
  for (int nt = 0; nt < 4; ++nt)
#pragma unroll
    for (int e = 0; e < 8; ++e) acc[nt][e] = 0.0f;

  float np = 0.0f;
#pragma unroll
  for (int ks = 0; ks < 2; ++ks) {
    v16bf a = load_a_frag(qp + qrow * M_ + ks * 32, M_);
#pragma unroll
    for (int t = 0; t < 8; ++t) {
      np += (float)a[t] * ksum[(size_t)bh * M_ + ks * 32 + hi * 8 + t];
      np += (float)a[t + 8] * ksum[(size_t)bh * M_ + ks * 32 + 16 + hi * 8 + t];
    }
#pragma unroll
    for (int nt = 0; nt < 4; ++nt) {
      v16bf bfr = load_b_frag(
          kvT + (size_t)bh * M_ * HD_ + (size_t)(nt * 16) * M_ + ks * 32, M_);
      acc[nt] = wmma_bf16(a, bfr, acc[nt]);
    }
  }
  np += __shfl_xor(np, 16, 32);     // combine the two K-halves of row cn
  if (hi == 0) norm[wid * 16 + cn] = np;
  __syncthreads();

#pragma unroll
  for (int nt = 0; nt < 4; ++nt)
#pragma unroll
    for (int j = 0; j < 8; ++j) {
      int row = hi * 8 + j;
      float v = acc[nt][j] / (norm[wid * 16 + row] + EPS_);
      size_t orow = (size_t)b * S_ + (size_t)st * 64 + wid * 16 + row;
      ctx[orow * H_ + h * HD_ + nt * 16 + cn] = (bf16)v;
    }
}

// ---------------------------------------------------------------------------
// Host launch
// ---------------------------------------------------------------------------
extern "C" void kernel_launch(void* const* d_in, const int* in_sizes, int n_in,
                              void* d_out, int out_size, void* d_ws,
                              size_t ws_size, hipStream_t stream) {
  (void)in_sizes; (void)n_in; (void)out_size; (void)ws_size;
  const float* X  = (const float*)d_in[0];
  const float* Wq = (const float*)d_in[1];
  const float* bq = (const float*)d_in[2];
  const float* Wk = (const float*)d_in[3];
  const float* bk = (const float*)d_in[4];
  const float* Wv = (const float*)d_in[5];
  const float* bv = (const float*)d_in[6];
  const float* Wo = (const float*)d_in[7];
  const float* bo = (const float*)d_in[8];
  const float* P  = (const float*)d_in[9];

  char* ws = (char*)d_ws;
  size_t o = 0;
  auto alloc = [&](size_t bytes) {
    size_t r = o;
    o += (bytes + 255) & ~(size_t)255;
    return r;
  };
  bf16*  X16    = (bf16*)(ws + alloc((size_t)BS * H_ * 2));   // reused as ctx16
  bf16*  Y16    = (bf16*)(ws + alloc((size_t)BS * N3 * 2));
  bf16*  WqkvT  = (bf16*)(ws + alloc((size_t)N3 * H_ * 2));
  float* bqkv   = (float*)(ws + alloc((size_t)N3 * 4));
  bf16*  WoT    = (bf16*)(ws + alloc((size_t)H_ * H_ * 2));
  bf16*  PT16   = (bf16*)(ws + alloc((size_t)NH_ * M_ * HD_ * 2));
  bf16*  qp     = (bf16*)(ws + alloc((size_t)B_ * NH_ * S_ * M_ * 2));
  bf16*  kpT    = (bf16*)(ws + alloc((size_t)B_ * NH_ * M_ * S_ * 2));
  bf16*  vT     = (bf16*)(ws + alloc((size_t)B_ * NH_ * HD_ * S_ * 2));
  float* ksum   = (float*)(ws + alloc((size_t)B_ * NH_ * M_ * 4));
  float* kvac   = (float*)(ws + alloc((size_t)B_ * NH_ * M_ * HD_ * 4));
  bf16*  kvT16  = (bf16*)(ws + alloc((size_t)B_ * NH_ * M_ * HD_ * 2));

  // 1) zero atomic accumulators (fresh every call)
  zero_kernel<<<64, 256, 0, stream>>>(ksum, (size_t)B_ * NH_ * M_);
  zero_kernel<<<768, 256, 0, stream>>>(kvac, (size_t)B_ * NH_ * M_ * HD_);
  // 2) conversions / transposed packs
  cvt_kernel<<<4096, 256, 0, stream>>>(X, X16, (size_t)BS * H_);
  cvtT768_kernel<<<((size_t)H_ * H_ + 255) / 256, 256, 0, stream>>>(Wo, WoT);
  cvtPT_kernel<<<((size_t)NH_ * M_ * HD_ + 255) / 256, 256, 0, stream>>>(P, PT16);
  pack_wqkvT_kernel<<<((size_t)N3 * H_ + 255) / 256, 256, 0, stream>>>(
      Wq, Wk, Wv, bq, bk, bv, WqkvT, bqkv);
  // 3) fused QKV projection GEMM: [32768,768] @ [768,2304] + bias -> bf16
  gemm_kernel<true><<<dim3(N3 / 128, BS / 128), 256, 0, stream>>>(
      X16, WqkvT, bqkv, (void*)Y16, N3, H_);
  // 4) random-feature map + exp; qp row-major, kp transposed; ksum accumulate
  feature_kernel<<<dim3(B_ * NH_, S_ / 64), 128, 0, stream>>>(Y16, PT16, qp,
                                                              kpT, ksum);
  // 5) v transpose, then kv summary (kp^T @ v) with fp32 atomics; kv -> kvT bf16
  vT_kernel<<<dim3(B_ * NH_, S_ / 64), 128, 0, stream>>>(Y16, vT);
  kv_kernel<<<dim3(B_ * NH_, 8), 128, 0, stream>>>(kpT, vT, kvac);
  cvtKVT_kernel<<<768, 256, 0, stream>>>(kvac, kvT16);
  // 6) ctx = (qp @ kv) / (qp . k_sum + eps), bf16 (overwrites X16 region)
  ctx_kernel<<<dim3(B_ * NH_, S_ / 64), 128, 0, stream>>>(qp, kvT16, ksum, X16);
  // 7) output GEMM: ctx @ Wo + bo -> fp32 d_out
  gemm_kernel<false><<<dim3(H_ / 128, BS / 128), 256, 0, stream>>>(
      X16, WoT, bo, d_out, H_, H_);
}
